// FNO2d_58652073394781
// MI455X (gfx1250) — compile-verified, matrix operand order checked by reading, not run on hardware
//
#include <hip/hip_runtime.h>
#include <hip/hip_fp16.h>
#include <math.h>

// ---------------------------------------------------------------------------
// FNO2d for MI455X (gfx1250, wave32, WMMA).
// Truncated-mode DFTs are dense GEMMs on v_wmma_f32_16x16x32_f16 (f16 in,
// f32 accumulate). Spectral-weight streaming (524 MB fp32 over 5 layers) is
// the HBM roofline (~23us @ 23.3 TB/s) and is read with coalesced, LDS-staged,
// prefetched loads. Everything else lives in the 192 MB L2. Workspace ~86 MB.
// ---------------------------------------------------------------------------

typedef _Float16 h16;
typedef __attribute__((ext_vector_type(16))) _Float16 v16h;
typedef __attribute__((ext_vector_type(8)))  float    v8f;

#define PI2F 6.283185307179586f

static __device__ __forceinline__ v8f vzero8() {
  v8f a;
#pragma unroll
  for (int e = 0; e < 8; ++e) a[e] = 0.0f;
  return a;
}

static __device__ __forceinline__ float gelu_exact(float v) {
  return 0.5f * v * (1.0f + erff(v * 0.70710678118654752f));
}

// One 16x16 f32 tile: C += A(16xK) * B(KxN), A rows contiguous (lda halves),
// B stored transposed: Bt[n][k] (ldb halves). kTiles steps of K=32.
static __device__ __forceinline__ v8f wtile(const h16* __restrict__ A, int lda,
                                            const h16* __restrict__ Bt, int ldb,
                                            int kTiles, v8f acc) {
  const int lane = threadIdx.x & 31;
  const int r = lane & 15;
  const int hs = (lane >> 4) << 4;  // 0 or 16 halves
  for (int kt = 0; kt < kTiles; ++kt) {
    v16h a = *(const v16h*)(A + (size_t)r * lda + kt * 32 + hs);
    v16h b = *(const v16h*)(Bt + (size_t)r * ldb + kt * 32 + hs);
    acc = __builtin_amdgcn_wmma_f32_16x16x32_f16(false, a, false, b,
                                                 (short)0, acc, false, false);
  }
  return acc;
}

// C/D fragment mapping (16x16 f32, wave32): n = lane&15, m = 8*(lane>>4)+e.

// ---------------------------------------------------------------------------
// Table builder: f16 DFT factor matrices + f16 copies of dense weights.
// ---------------------------------------------------------------------------
__global__ void k_tables(h16* cyt, h16* a2, h16* a4, h16* e5t,
                         h16* wwh, h16* fc1t, h16* h1t,
                         const float* w_w, const float* fc1_w, const float* head1_w) {
  const int tid = blockIdx.x * blockDim.x + threadIdx.x;
  const int nth = gridDim.x * blockDim.x;
  // Cyt[n(48)][y(256)]: forward Y-DFT, cols packed [Re(20)|Im(20)|pad]
  for (int i = tid; i < 48 * 256; i += nth) {
    int n = i >> 8, y = i & 255;
    float v = 0.f;
    if (n < 20)       v =  cosf(PI2F * n * y / 256.0f);
    else if (n < 40)  v = -sinf(PI2F * (n - 20) * y / 256.0f);
    cyt[i] = (h16)v;
  }
  // A2[m(96)][k(512)]: forward X-DFT with complex combine. Rows 0..39 = Fr,
  // 40..79 = Fi, over kx freq {0..19, -20..-1}. k<256 hits Pr, k>=256 hits Pi.
  for (int i = tid; i < 96 * 512; i += nth) {
    int m = i >> 9, k = i & 511;
    float v = 0.f;
    if (m < 80) {
      int r = (m < 40) ? m : m - 40;
      float fx = (r < 20) ? (float)r : (float)(r - 40);
      float x = (float)(k & 255);
      float th = PI2F * fx * x / 256.0f;
      if (m < 40) v = (k < 256) ?  cosf(th) : sinf(th);
      else        v = (k < 256) ? -sinf(th) : cosf(th);
    }
    a2[i] = (h16)v;
  }
  // A4[x(256)][k(96)]: inverse X-DFT factors. k<40: cos, 40..79: sin, rest 0.
  for (int i = tid; i < 256 * 96; i += nth) {
    int x = i / 96, k = i - x * 96;
    float v = 0.f;
    if (k < 80) {
      int r = (k < 40) ? k : k - 40;
      float fx = (r < 20) ? (float)r : (float)(r - 40);
      float th = PI2F * fx * x / 256.0f;
      v = (k < 40) ? cosf(th) : sinf(th);
    }
    a4[i] = (h16)v;
  }
  // E5t[y(256)][k(64)]: inverse Y-DFT (transposed), 1/65536 + x2 folded in.
  for (int i = tid; i < 256 * 64; i += nth) {
    int y = i >> 6, k = i & 63;
    float v = 0.f;
    if (k < 20) {
      float c = (k == 0) ? 1.0f : 2.0f;
      v = c * cosf(PI2F * k * y / 256.0f) * (1.0f / 65536.0f);
    } else if (k < 40) {
      int kk = k - 20;
      float c = (kk == 0) ? 1.0f : 2.0f;
      v = -c * sinf(PI2F * kk * y / 256.0f) * (1.0f / 65536.0f);
    }
    e5t[i] = (h16)v;
  }
  // wwh[l][o][c] = w_w (already (5,o,c))
  for (int i = tid; i < 5 * 128 * 128; i += nth) wwh[i] = (h16)w_w[i];
  // fc1t[n][c] = fc1_w[c][n]
  for (int i = tid; i < 128 * 128; i += nth) {
    int n = i >> 7, c = i & 127;
    fc1t[i] = (h16)fc1_w[c * 128 + n];
  }
  // h1t[hd][n(64)][c(128)] = head1_w[hd][c][n]
  for (int i = tid; i < 4 * 64 * 128; i += nth) {
    int hd = i >> 13, n = (i >> 7) & 63, c = i & 127;
    h1t[i] = (h16)head1_w[((size_t)hd * 128 + c) * 64 + n];
  }
}

// ---------------------------------------------------------------------------
// Lift: h[b][c][x][y] (f16, 256x256 zero-padded) from x + coord grids.
// ---------------------------------------------------------------------------
__global__ void k_fc0(const float* __restrict__ xin, const float* __restrict__ w,
                      const float* __restrict__ bias, h16* __restrict__ h) {
  const int b = blockIdx.y, xi = blockIdx.x, y = threadIdx.x;
  const bool valid = (xi < 240) && (y < 240);
  float in[5] = {0.f, 0.f, 0.f, 0.f, 0.f};
  if (valid) {
    const float* p = xin + (((size_t)b * 240 + xi) * 240 + y) * 3;
    in[0] = p[0]; in[1] = p[1]; in[2] = p[2];
    in[3] = xi * (1.0f / 239.0f);
    in[4] = y * (1.0f / 239.0f);
  }
  for (int c = 0; c < 128; ++c) {
    float acc = 0.f;
    if (valid) {
      acc = bias[c];
#pragma unroll
      for (int j = 0; j < 5; ++j) acc += in[j] * w[j * 128 + c];
    }
    h[(((size_t)b * 128 + c) * 256 + xi) * 256 + y] = (h16)acc;
  }
}

// ---------------------------------------------------------------------------
// Step A (Y-DFT): Pt[bc][n(40)][x] = (h(64x256) * Cy(256x40))^T per x-tile.
// grid = (xtile 4, bc 256)
// ---------------------------------------------------------------------------
__global__ void k_ydft(const h16* __restrict__ h, const h16* __restrict__ cyt,
                       h16* __restrict__ pt) {
  __shared__ __align__(32) h16 lA[64 * 256];  // 32 KB
  __shared__ __align__(32) h16 lB[48 * 256];  // 24 KB
  const int xt = blockIdx.x, bc = blockIdx.y;
  {
    const float4* s = (const float4*)(h + ((size_t)bc * 256 + xt * 64) * 256);
    float4* d = (float4*)lA;
    for (int i = threadIdx.x; i < 2048; i += 256) d[i] = s[i];
    const float4* s2 = (const float4*)cyt;
    float4* d2 = (float4*)lB;
    for (int i = threadIdx.x; i < 1536; i += 256) d2[i] = s2[i];
  }
  __syncthreads();
  const int wave = threadIdx.x >> 5, lane = threadIdx.x & 31;
  for (int t = wave; t < 12; t += 8) {   // 4 m-tiles x 3 n-tiles
    int mt = t / 3, nt = t - mt * 3;
    v8f acc = vzero8();
    acc = wtile(lA + mt * 16 * 256, 256, lB + nt * 16 * 256, 256, 8, acc);
    int n = nt * 16 + (lane & 15);
    if (n < 40) {
#pragma unroll
      for (int e = 0; e < 8; ++e) {
        int m = mt * 16 + ((lane >> 4) * 8 + e);
        pt[((size_t)bc * 40 + n) * 256 + (xt * 64 + m)] = (h16)acc[e];
      }
    }
  }
}

// ---------------------------------------------------------------------------
// Step B (X-DFT): per-(b,c) F(80x20) = A2(80x512) * [Pr;Pi](512x20).
// A2 is constant -> streamed from L2. grid = 256
// ---------------------------------------------------------------------------
__global__ void k_xdft(const h16* __restrict__ pt, const h16* __restrict__ a2,
                       float* __restrict__ F) {
  __shared__ __align__(32) h16 lB[32 * 512];  // 32 KB  (Bt[n][k])
  const int bc = blockIdx.x;
  // rows n<20: [Pr row n | Pi row n] (each 256 halves, contiguous in pt)
  for (int j = threadIdx.x; j < 20 * 64; j += 256) {
    int n = j >> 6, q = j & 63;
    int row = (q < 32) ? n : 20 + n;
    ((float4*)(lB + n * 512))[q] =
        ((const float4*)(pt + ((size_t)bc * 40 + row) * 256))[q & 31];
  }
  for (int j = threadIdx.x; j < 12 * 64; j += 256) {
    int n = 20 + (j >> 6), q = j & 63;
    float4 z; z.x = z.y = z.z = z.w = 0.f;
    ((float4*)(lB + n * 512))[q] = z;
  }
  __syncthreads();
  const int wave = threadIdx.x >> 5, lane = threadIdx.x & 31;
  for (int t = wave; t < 12; t += 8) {   // 6 m-tiles x 2 n-tiles
    int mt = t >> 1, nt = t & 1;
    v8f acc = vzero8();
    acc = wtile(a2 + (size_t)mt * 16 * 512, 512, lB + nt * 16 * 512, 512, 16, acc);
    int n = nt * 16 + (lane & 15);
    if (n < 20) {
#pragma unroll
      for (int e = 0; e < 8; ++e) {
        int m = mt * 16 + ((lane >> 4) * 8 + e);
        if (m < 80) F[((size_t)bc * 80 + m) * 20 + n] = acc[e];
      }
    }
  }
}

// ---------------------------------------------------------------------------
// Mode mix: per mode (kx,ky): O[b][o] = sum_i F[b][i] * w[i][o] (complex).
// HBM-bound stage: 105 MB fp32 weights per layer. One WG per (kx, 8-wide
// o-block); the contiguous 40-float (my,re/im) runs are staged into LDS with
// coalesced float2 loads + global_prefetch of the next i-block. Each thread
// owns 2 (o,b,my) accumulators. Output is written as the complex-as-real 2x2
// block Bt consumed by the inverse X-DFT GEMM. grid = (40, 16)
// ---------------------------------------------------------------------------
__global__ void k_modemix(const float* __restrict__ F, const float* __restrict__ w1,
                          const float* __restrict__ w2, h16* __restrict__ b4,
                          int layer) {
  __shared__ float fr[2][128][20];   // 20 KB
  __shared__ float fi[2][128][20];   // 20 KB
  __shared__ float wbuf[4 * 8 * 40]; // 5 KB
  const int mx = blockIdx.x;
  const int o0 = blockIdx.y * 8;
  // Stage F[b][i][my] for this kx (re/im planes).
  for (int j = threadIdx.x; j < 2 * 128 * 20; j += 256) {
    int bq = j / (128 * 20);
    int r = j - bq * 128 * 20;
    int i = r / 20, my = r - i * 20;
    size_t bc = (size_t)bq * 128 + i;
    fr[bq][i][my] = F[(bc * 80 + mx) * 20 + my];
    fi[bq][i][my] = F[(bc * 80 + 40 + mx) * 20 + my];
  }
  const float* w;
  int wmx;
  if (mx < 20) { w = w1; wmx = mx; } else { w = w2; wmx = mx - 20; }
  // Combos c = (o_l*2 + b)*20 + my, 8*2*20 = 320 per WG; 2 per thread.
  const int c0 = threadIdx.x, c1 = threadIdx.x + 256;
  const int ol0 = c0 / 40, b0 = (c0 / 20) & 1, my0 = c0 % 20;
  const int ol1 = c1 / 40, b1 = (c1 / 20) & 1, my1 = c1 % 20;
  float sr0 = 0.f, si0 = 0.f, sr1 = 0.f, si1 = 0.f;
  __syncthreads();
  for (int i0 = 0; i0 < 128; i0 += 4) {
    // Stage w[i0..i0+3][o0..o0+7][mx][0..19][2] -> LDS (coalesced float2).
    for (int j = threadIdx.x; j < 4 * 8 * 20; j += 256) {
      int ii = j / 160;
      int r = j - ii * 160;
      int ol = r / 20, p = r - ol * 20;
      const float* src =
          w + (((size_t)(layer * 128 + i0 + ii) * 128 + (o0 + ol)) * 400 +
               (size_t)wmx * 20) * 2;
      ((float2*)wbuf)[j] = ((const float2*)src)[p];
      if (i0 + 4 < 128)
        __builtin_prefetch(src + 4 * (size_t)128 * 800 + p * 2, 0, 0);
    }
    __syncthreads();
#pragma unroll
    for (int ii = 0; ii < 4; ++ii) {
      const int i = i0 + ii;
      {
        float ar = fr[b0][i][my0], ai = fi[b0][i][my0];
        float wr = wbuf[(ii * 8 + ol0) * 40 + 2 * my0];
        float wi = wbuf[(ii * 8 + ol0) * 40 + 2 * my0 + 1];
        sr0 += ar * wr - ai * wi;
        si0 += ar * wi + ai * wr;
      }
      if (c1 < 320) {
        float ar = fr[b1][i][my1], ai = fi[b1][i][my1];
        float wr = wbuf[(ii * 8 + ol1) * 40 + 2 * my1];
        float wi = wbuf[(ii * 8 + ol1) * 40 + 2 * my1 + 1];
        sr1 += ar * wr - ai * wi;
        si1 += ar * wi + ai * wr;
      }
    }
    __syncthreads();
  }
  {
    h16* B = b4 + (size_t)(b0 * 128 + o0 + ol0) * 48 * 96;
    B[my0 * 96 + mx]             = (h16)sr0;
    B[(20 + my0) * 96 + mx]      = (h16)si0;
    B[my0 * 96 + 40 + mx]        = (h16)(-si0);
    B[(20 + my0) * 96 + 40 + mx] = (h16)sr0;
  }
  if (c1 < 320) {
    h16* B = b4 + (size_t)(b1 * 128 + o0 + ol1) * 48 * 96;
    B[my1 * 96 + mx]             = (h16)sr1;
    B[(20 + my1) * 96 + mx]      = (h16)si1;
    B[my1 * 96 + 40 + mx]        = (h16)(-si1);
    B[(20 + my1) * 96 + 40 + mx] = (h16)sr1;
  }
}

// ---------------------------------------------------------------------------
// Inverse X-DFT: per-(b,o) T(256x40) = A4(256x96) * B4(96x40). grid = 256
// ---------------------------------------------------------------------------
__global__ void k_invx(const h16* __restrict__ b4, const h16* __restrict__ a4,
                       h16* __restrict__ T) {
  __shared__ __align__(32) h16 lA[256 * 96];  // 48 KB
  __shared__ __align__(32) h16 lB[48 * 96];   // 9 KB
  const int bo = blockIdx.x;
  {
    const float4* s = (const float4*)a4;
    float4* d = (float4*)lA;
    for (int i = threadIdx.x; i < 3072; i += 256) d[i] = s[i];
    const float4* s2 = (const float4*)(b4 + (size_t)bo * 48 * 96);
    float4* d2 = (float4*)lB;
    for (int i = threadIdx.x; i < 576; i += 256) d2[i] = s2[i];
  }
  __syncthreads();
  const int wave = threadIdx.x >> 5, lane = threadIdx.x & 31;
  for (int t = wave; t < 48; t += 8) {   // 16 m-tiles x 3 n-tiles
    int mt = t / 3, nt = t - mt * 3;
    v8f acc = vzero8();
    acc = wtile(lA + mt * 16 * 96, 96, lB + nt * 16 * 96, 96, 3, acc);
    int n = nt * 16 + (lane & 15);
    if (n < 40) {
#pragma unroll
      for (int e = 0; e < 8; ++e) {
        int m = mt * 16 + ((lane >> 4) * 8 + e);
        T[((size_t)bo * 256 + m) * 64 + n] = (h16)acc[e];
      }
    }
  }
}

// ---------------------------------------------------------------------------
// Fused inverse Y-DFT + 1x1 conv + bias + GELU:
//   h'[b][o][x][y] = act( T(16x64)*E(64x256) + Ww(16x128)*h(128x256) + b[o] )
// grid = (ot*2+yhalf : 16, x : 256, b : 2)
// ---------------------------------------------------------------------------
__global__ void k_spec_lin(const h16* __restrict__ hin, h16* __restrict__ hout,
                           const h16* __restrict__ T, const h16* __restrict__ e5t,
                           const h16* __restrict__ wwh, const float* __restrict__ wb,
                           int layer, int do_gelu) {
  __shared__ __align__(32) h16 lH[128 * 128];  // 32 KB  Bt[y][c]
  __shared__ __align__(32) h16 lT[16 * 64];    // 2 KB   A rows (o)
  __shared__ __align__(32) h16 lW[16 * 128];   // 4 KB   A rows (o)
  const int ot = blockIdx.x >> 1, yh = blockIdx.x & 1;
  const int xi = blockIdx.y, b = blockIdx.z;
  const int y0 = yh * 128, o0 = ot * 16;
  // Bt[y][c] gather: vectorized along contiguous y (float4 = 8 halves).
  for (int j = threadIdx.x; j < 2048; j += 256) {
    int c = j >> 4, q = j & 15;
    float4 v = ((const float4*)(hin + (((size_t)b * 128 + c) * 256 + xi) * 256 +
                                y0))[q];
    const h16* hv = (const h16*)&v;
#pragma unroll
    for (int jj = 0; jj < 8; ++jj) lH[(q * 8 + jj) * 128 + c] = hv[jj];
  }
  for (int i = threadIdx.x; i < 128; i += 256) {  // 16 rows x 8 float4
    int m = i >> 3, q = i & 7;
    ((float4*)lT)[i] =
        ((const float4*)(T + (((size_t)b * 128 + o0 + m) * 256 + xi) * 64))[q];
  }
  for (int i = threadIdx.x; i < 256; i += 256) {  // 16 rows x 16 float4
    int m = i >> 4, q = i & 15;
    ((float4*)lW)[i] =
        ((const float4*)(wwh + ((size_t)layer * 128 + o0 + m) * 128))[q];
  }
  __syncthreads();
  const int wave = threadIdx.x >> 5, lane = threadIdx.x & 31;
  for (int nt = wave; nt < 8; nt += 8) {          // one 16-col tile per wave
    v8f acc = vzero8();
    acc = wtile(lT, 64, e5t + (size_t)(y0 + nt * 16) * 64, 64, 2, acc);   // spectral
    acc = wtile(lW, 128, lH + nt * 16 * 128, 128, 4, acc);                // 1x1 conv
    int n = y0 + nt * 16 + (lane & 15);
#pragma unroll
    for (int e = 0; e < 8; ++e) {
      int o = o0 + (lane >> 4) * 8 + e;
      float v = acc[e] + wb[layer * 128 + o];
      if (do_gelu) v = gelu_exact(v);
      hout[(((size_t)b * 128 + o) * 256 + xi) * 256 + n] = (h16)v;
    }
  }
}

// ---------------------------------------------------------------------------
// Fused crop + fc1 + GELU + 4 heads (128->64 GELU, 64->1).
// grid = (ytile 15, x 240, b 2); 16 pixels per workgroup.
// ---------------------------------------------------------------------------
__global__ void k_head(const h16* __restrict__ hin, const h16* __restrict__ fc1t,
                       const float* __restrict__ fc1_b, const h16* __restrict__ h1t,
                       const float* __restrict__ h1b, const float* __restrict__ h2w,
                       const float* __restrict__ h2b, float* __restrict__ out) {
  __shared__ __align__(32) h16 lA[16 * 128];      // pixels x channels
  __shared__ __align__(32) h16 lG[16 * 128];      // fc1 activations
  __shared__ __align__(32) h16 lO[4 * 16 * 64];   // head1 activations
  const int yt = blockIdx.x, xi = blockIdx.y, b = blockIdx.z;
  const int y0 = yt * 16;
  for (int j = threadIdx.x; j < 256; j += 256) {  // vectorized transpose gather
    int c = j >> 1, q = j & 1;
    float4 v = ((const float4*)(hin + (((size_t)b * 128 + c) * 256 + xi) * 256 +
                                y0))[q];
    const h16* hv = (const h16*)&v;
#pragma unroll
    for (int jj = 0; jj < 8; ++jj) lA[(q * 8 + jj) * 128 + c] = hv[jj];
  }
  __syncthreads();
  const int wave = threadIdx.x >> 5, lane = threadIdx.x & 31;
  for (int nt = wave; nt < 8; nt += 8) {          // fc1: 8 col tiles
    v8f acc = vzero8();
    acc = wtile(lA, 128, fc1t + (size_t)nt * 16 * 128, 128, 4, acc);
    int n = nt * 16 + (lane & 15);
#pragma unroll
    for (int e = 0; e < 8; ++e) {
      int m = (lane >> 4) * 8 + e;
      lG[m * 128 + n] = (h16)gelu_exact(acc[e] + fc1_b[n]);
    }
  }
  __syncthreads();
  for (int t = wave; t < 16; t += 8) {            // 4 heads x 4 col tiles
    int hd = t >> 2, nt = t & 3;
    v8f acc = vzero8();
    acc = wtile(lG, 128, h1t + ((size_t)hd * 64 + nt * 16) * 128, 128, 4, acc);
    int n = nt * 16 + (lane & 15);
#pragma unroll
    for (int e = 0; e < 8; ++e) {
      int m = (lane >> 4) * 8 + e;
      lO[(hd * 16 + m) * 64 + n] = (h16)gelu_exact(acc[e] + h1b[hd * 64 + n]);
    }
  }
  __syncthreads();
  if (threadIdx.x < 64) {                          // head2: 64->1 dots
    int hd = threadIdx.x >> 4, m = threadIdx.x & 15;
    float acc = h2b[hd];
    for (int d = 0; d < 64; ++d)
      acc += (float)lO[(hd * 16 + m) * 64 + d] * h2w[hd * 64 + d];
    out[(((size_t)hd * 2 + b) * 240 + xi) * 240 + (y0 + m)] = acc;
  }
}

// ---------------------------------------------------------------------------
extern "C" void kernel_launch(void* const* d_in, const int* in_sizes, int n_in,
                              void* d_out, int out_size, void* d_ws, size_t ws_size,
                              hipStream_t stream) {
  (void)in_sizes; (void)n_in; (void)out_size; (void)ws_size;
  const float* x       = (const float*)d_in[0];
  const float* fc0_w   = (const float*)d_in[1];
  const float* fc0_b   = (const float*)d_in[2];
  const float* spec_w1 = (const float*)d_in[3];
  const float* spec_w2 = (const float*)d_in[4];
  const float* w_w     = (const float*)d_in[5];
  const float* w_b     = (const float*)d_in[6];
  const float* fc1_w   = (const float*)d_in[7];
  const float* fc1_b   = (const float*)d_in[8];
  const float* head1_w = (const float*)d_in[9];
  const float* head1_b = (const float*)d_in[10];
  const float* head2_w = (const float*)d_in[11];
  const float* head2_b = (const float*)d_in[12];

  char* wp = (char*)d_ws;
  auto carve = [&](size_t bytes) {
    void* p = (void*)wp;
    wp += (bytes + 255) & ~(size_t)255;
    return p;
  };
  const size_t HBYTES  = (size_t)2 * 128 * 256 * 256 * 2;   // 33.5 MB each
  const size_t B4BYTES = (size_t)256 * 48 * 96 * 2;
  const size_t TBYTES  = (size_t)256 * 256 * 64 * 2;
  h16*   hA   = (h16*)carve(HBYTES);
  h16*   hB   = (h16*)carve(HBYTES);
  h16*   pt   = (h16*)carve((size_t)256 * 40 * 256 * 2);
  float* F    = (float*)carve((size_t)256 * 80 * 20 * 4);
  h16*   b4   = (h16*)carve(B4BYTES);
  h16*   T    = (h16*)carve(TBYTES);
  h16*   cyt  = (h16*)carve((size_t)48 * 256 * 2);
  h16*   a2   = (h16*)carve((size_t)96 * 512 * 2);
  h16*   a4   = (h16*)carve((size_t)256 * 96 * 2);
  h16*   e5t  = (h16*)carve((size_t)256 * 64 * 2);
  h16*   wwh  = (h16*)carve((size_t)5 * 128 * 128 * 2);
  h16*   fc1t = (h16*)carve((size_t)128 * 128 * 2);
  h16*   h1t  = (h16*)carve((size_t)4 * 64 * 128 * 2);

  // Zero padded tile regions once per launch (graph-capturable).
  hipMemsetAsync(b4, 0, B4BYTES, stream);
  hipMemsetAsync(T, 0, TBYTES, stream);

  k_tables<<<64, 256, 0, stream>>>(cyt, a2, a4, e5t, wwh, fc1t, h1t,
                                   w_w, fc1_w, head1_w);
  k_fc0<<<dim3(256, 2), 256, 0, stream>>>(x, fc0_w, fc0_b, hA);

  h16* cur = hA;
  h16* nxt = hB;
  for (int l = 0; l < 5; ++l) {
    k_ydft<<<dim3(4, 256), 256, 0, stream>>>(cur, cyt, pt);
    k_xdft<<<256, 256, 0, stream>>>(pt, a2, F);
    k_modemix<<<dim3(40, 16), 256, 0, stream>>>(F, spec_w1, spec_w2, b4, l);
    k_invx<<<256, 256, 0, stream>>>(b4, a4, T);
    k_spec_lin<<<dim3(16, 256, 2), 256, 0, stream>>>(cur, nxt, T, e5t, wwh,
                                                     w_b, l, (l < 4) ? 1 : 0);
    h16* tmp = cur; cur = nxt; nxt = tmp;
  }
  k_head<<<dim3(15, 240, 2), 256, 0, stream>>>(cur, fc1t, fc1_b, h1t, head1_b,
                                               head2_w, head2_b, (float*)d_out);
}
